// Encoder_GNN_u_weighted_46815143526426
// MI455X (gfx1250) — matrix operand order repaired and checked by reading, not compiled, not run on hardware
//
#include <hip/hip_runtime.h>
#include <hip/hip_bf16.h>

#define N_M 100000
#define N_D 50000
#define NE  800000
#define DD  128
#define HH  256
#define OO  128

typedef __bf16 bf16_t;
typedef __attribute__((ext_vector_type(4)))  __bf16 v4bf;
typedef __attribute__((ext_vector_type(8)))  __bf16 v8bf;
typedef __attribute__((ext_vector_type(16))) __bf16 v16bf;
typedef __attribute__((ext_vector_type(8)))  float  v8f;

// ---------------- elementwise helpers ----------------

__global__ void k_zero_f32(float* __restrict__ p, int n4) {
    int i = blockIdx.x * blockDim.x + threadIdx.x;
    if (i < n4) ((float4*)p)[i] = make_float4(0.f, 0.f, 0.f, 0.f);
}

__global__ void k_cvt_bf16(const float* __restrict__ in, bf16_t* __restrict__ out, int n4) {
    int i = blockIdx.x * blockDim.x + threadIdx.x;
    if (i >= n4) return;
    float4 v = ((const float4*)in)[i];
    v4bf o;
    o[0] = (bf16_t)v.x; o[1] = (bf16_t)v.y; o[2] = (bf16_t)v.z; o[3] = (bf16_t)v.w;
    ((v4bf*)out)[i] = o;
}

// W: [K,H] f32 row-major  ->  Wt: [H,K] bf16 row-major (columns become contiguous)
__global__ void k_transpose_cvt(const float* __restrict__ W, bf16_t* __restrict__ Wt,
                                int K, int H) {
    int i = blockIdx.x * blockDim.x + threadIdx.x;
    if (i >= K * H) return;
    int k = i / H, h = i - k * H;
    Wt[h * K + k] = (bf16_t)W[i];
}

// ---------------- edge scatter-add (segment_sum) ----------------

template <bool WEIGHTED, int DIM>
__global__ void k_scatter_add(const float* __restrict__ x, const int* __restrict__ src,
                              const int* __restrict__ dst, const float* __restrict__ w,
                              float* __restrict__ agg) {
    const int TPE = DIM / 4;                      // threads per edge, float4 each
    long tid = (long)blockIdx.x * blockDim.x + threadIdx.x;
    long e = tid / TPE;
    if (e >= NE) return;
    int c = (int)(tid - e * TPE) * 4;
    int s = src[e], d = dst[e];
    float4 v = *(const float4*)(x + (long)s * DIM + c);
    float* p = agg + (long)d * DIM + c;
    if (WEIGHTED) {
        float wt = w[e];
        atomicAdd(p + 0, v.x * wt); atomicAdd(p + 1, v.y * wt);
        atomicAdd(p + 2, v.z * wt); atomicAdd(p + 3, v.w * wt);
    } else {
        atomicAdd(p + 0, v.x); atomicAdd(p + 1, v.y);
        atomicAdd(p + 2, v.z); atomicAdd(p + 3, v.w);
    }
}

// ---------------- fused WMMA GEMM: out = act(A0@W0^T' + [A1@W1^T'] + bias) ----------------
// A*: [N,K] bf16 row-major. W*: [H,K] bf16 (pre-transposed).
// One wave computes a 16x64 strip (4 adjacent 16x16 N-tiles): the A fragment is
// loaded once per K-step and reused across 4 WMMAs -> 4x less A read traffic.

template <bool TWO, bool RELU, bool OUTBF>
__global__ __launch_bounds__(256)
void k_gemm_wmma(const bf16_t* __restrict__ A0, const bf16_t* __restrict__ W0,
                 const bf16_t* __restrict__ A1, const bf16_t* __restrict__ W1,
                 const float* __restrict__ bias, float* __restrict__ outf,
                 bf16_t* __restrict__ outb, int N, int K, int H) {
    constexpr int NT = 4;                         // n-tiles per wave (strip = 64 cols)
    int wave   = blockIdx.x * (blockDim.x >> 5) + (threadIdx.x >> 5);
    int lane   = threadIdx.x & 31;
    int grpCnt = H >> 6;                          // strips per row block
    int tm     = wave / grpCnt;
    int grp    = wave - tm * grpCnt;
    if (tm >= (N >> 4)) return;                   // wave-uniform: EXEC stays all-1s

    int half = lane >> 4;                         // 0: lanes 0-15, 1: lanes 16-31
    int lr   = lane & 15;
    // A fragment: row tm*16+lr; lanes 0-15 hold K={k0..k0+7, k0+16..k0+23},
    //             lanes 16-31 hold K={k0+8..k0+15, k0+24..k0+31}  (ISA 16-bit A layout)
    long abase = (long)(tm * 16 + lr) * K + half * 8;
    // B fragment: col grp*64 + j*16 + lr; same K split as A (lanes 0-15 K={k0..k0+7, k0+16..k0+23}, lanes 16-31 +8)
    long bbase = (long)(grp * 64 + lr) * K + half * 8;
    long btile = (long)16 * K;                    // bytes/2 between adjacent n-tiles

    v8f acc[NT];
#pragma unroll
    for (int j = 0; j < NT; ++j) acc[j] = (v8f){0.f, 0.f, 0.f, 0.f, 0.f, 0.f, 0.f, 0.f};

    for (int pass = 0; pass < (TWO ? 2 : 1); ++pass) {
        const bf16_t* A = (TWO && pass) ? A1 : A0;
        const bf16_t* W = (TWO && pass) ? W1 : W0;
        for (int k0 = 0; k0 < K; k0 += 32) {
            v8bf alo = *(const v8bf*)(A + abase + k0);
            v8bf ahi = *(const v8bf*)(A + abase + k0 + 16);
            v16bf a = __builtin_shufflevector(alo, ahi, 0,1,2,3,4,5,6,7,8,9,10,11,12,13,14,15);
#pragma unroll
            for (int j = 0; j < NT; ++j) {
                const bf16_t* wp = W + bbase + j * btile + k0;
                v8bf blo = *(const v8bf*)(wp);
                v8bf bhi = *(const v8bf*)(wp + 16);
                v16bf b = __builtin_shufflevector(blo, bhi, 0,1,2,3,4,5,6,7,8,9,10,11,12,13,14,15);
                acc[j] = __builtin_amdgcn_wmma_f32_16x16x32_bf16(
                    /*neg_a=*/false, a, /*neg_b=*/false, b,
                    /*c_mod=*/(short)0, acc[j], /*reuse_a=*/false, /*reuse_b=*/false);
            }
        }
    }

    // D layout: VGPR r -> (M = half*8 + r, N = lr)
    long rowbase = (long)(tm * 16 + half * 8);
#pragma unroll
    for (int j = 0; j < NT; ++j) {
        int col = grp * 64 + j * 16 + lr;
        float bv = bias[col];
#pragma unroll
        for (int r = 0; r < 8; ++r) {
            float v = acc[j][r] + bv;
            if (RELU) v = fmaxf(v, 0.f);
            long idx = (rowbase + r) * H + col;
            if (OUTBF) outb[idx] = (bf16_t)v;
            else       outf[idx] = v;
        }
    }
}

// ---------------- driver ----------------

extern "C" void kernel_launch(void* const* d_in, const int* in_sizes, int n_in,
                              void* d_out, int out_size, void* d_ws, size_t ws_size,
                              hipStream_t stream) {
    (void)in_sizes; (void)n_in; (void)out_size; (void)ws_size;

    const float* x_m     = (const float*)d_in[0];
    const float* x_d     = (const float*)d_in[1];
    const int*   mm_src  = (const int*)d_in[2];
    const int*   mm_dst  = (const int*)d_in[3];
    const int*   md_src  = (const int*)d_in[4];
    const int*   md_dst  = (const int*)d_in[5];
    const float* ew      = (const float*)d_in[6];
    const float* W_rel1  = (const float*)d_in[7];
    const float* b_rel1  = (const float*)d_in[8];
    const float* W_root1 = (const float*)d_in[9];
    const float* W_rel2  = (const float*)d_in[10];
    const float* b_rel2  = (const float*)d_in[11];
    const float* W_root2 = (const float*)d_in[12];
    const float* W_rel3  = (const float*)d_in[13];
    const float* b_rel3  = (const float*)d_in[14];
    const float* W_root3 = (const float*)d_in[15];
    const float* W_lin   = (const float*)d_in[16];
    const float* b_lin   = (const float*)d_in[17];

    char* ws = (char*)d_ws;
    size_t off = 0;
    auto take = [&](size_t bytes) -> char* {
        char* p = ws + off;
        off = (off + bytes + 255) & ~(size_t)255;
        return p;
    };
    float*  agg1   = (float*)take((size_t)N_M * DD * 4);
    float*  agg2   = (float*)take((size_t)N_D * DD * 4);
    float*  agg3   = (float*)take((size_t)N_D * HH * 4);
    float*  movie  = (float*)take((size_t)N_M * HH * 4);
    float*  userx  = (float*)take((size_t)N_D * HH * 4);
    bf16_t* xmB    = (bf16_t*)take((size_t)N_M * DD * 2);
    bf16_t* xdB    = (bf16_t*)take((size_t)N_D * DD * 2);
    bf16_t* agg1B  = (bf16_t*)take((size_t)N_M * DD * 2);
    bf16_t* agg2B  = (bf16_t*)take((size_t)N_D * DD * 2);
    bf16_t* agg3B  = (bf16_t*)take((size_t)N_D * HH * 2);
    bf16_t* userxB = (bf16_t*)take((size_t)N_D * HH * 2);
    bf16_t* ux2B   = (bf16_t*)take((size_t)N_D * HH * 2);
    bf16_t* Wr1T   = (bf16_t*)take((size_t)DD * HH * 2);
    bf16_t* Wo1T   = (bf16_t*)take((size_t)DD * HH * 2);
    bf16_t* Wr2T   = (bf16_t*)take((size_t)DD * HH * 2);
    bf16_t* Wo2T   = (bf16_t*)take((size_t)DD * HH * 2);
    bf16_t* Wr3T   = (bf16_t*)take((size_t)HH * HH * 2);
    bf16_t* Wo3T   = (bf16_t*)take((size_t)HH * HH * 2);
    bf16_t* WlT    = (bf16_t*)take((size_t)HH * OO * 2);

    const int T = 256;
    auto blk = [](long n) { return (int)((n + 255) / 256); };

    // 1) zero accumulators (must happen every call)
    k_zero_f32<<<blk((long)N_M * DD / 4), T, 0, stream>>>(agg1, N_M * DD / 4);
    k_zero_f32<<<blk((long)N_D * DD / 4), T, 0, stream>>>(agg2, N_D * DD / 4);
    k_zero_f32<<<blk((long)N_D * HH / 4), T, 0, stream>>>(agg3, N_D * HH / 4);

    // 2) weights -> transposed bf16
    k_transpose_cvt<<<blk(DD * HH), T, 0, stream>>>(W_rel1,  Wr1T, DD, HH);
    k_transpose_cvt<<<blk(DD * HH), T, 0, stream>>>(W_root1, Wo1T, DD, HH);
    k_transpose_cvt<<<blk(DD * HH), T, 0, stream>>>(W_rel2,  Wr2T, DD, HH);
    k_transpose_cvt<<<blk(DD * HH), T, 0, stream>>>(W_root2, Wo2T, DD, HH);
    k_transpose_cvt<<<blk(HH * HH), T, 0, stream>>>(W_rel3,  Wr3T, HH, HH);
    k_transpose_cvt<<<blk(HH * HH), T, 0, stream>>>(W_root3, Wo3T, HH, HH);
    k_transpose_cvt<<<blk(HH * OO), T, 0, stream>>>(W_lin,   WlT,  HH, OO);

    // 3) activations -> bf16
    k_cvt_bf16<<<blk((long)N_M * DD / 4), T, 0, stream>>>(x_m, xmB, N_M * DD / 4);
    k_cvt_bf16<<<blk((long)N_D * DD / 4), T, 0, stream>>>(x_d, xdB, N_D * DD / 4);

    // 4) aggregations in the *input* dim (linearity: segsum(x)@W == segsum(x)@W)
    k_scatter_add<true,  DD><<<blk((long)NE * (DD / 4)), T, 0, stream>>>(x_m, mm_src, mm_dst, ew,      agg1);
    k_scatter_add<false, DD><<<blk((long)NE * (DD / 4)), T, 0, stream>>>(x_m, md_src, md_dst, nullptr, agg2);
    k_cvt_bf16<<<blk((long)N_M * DD / 4), T, 0, stream>>>(agg1, agg1B, N_M * DD / 4);
    k_cvt_bf16<<<blk((long)N_D * DD / 4), T, 0, stream>>>(agg2, agg2B, N_D * DD / 4);

    // 5) conv1: movie = ReLU(agg1@Wrel1 + x_m@Wroot1 + b1)   [N_M x 256]
    {
        long waves = (long)(N_M / 16) * (HH / 64);
        k_gemm_wmma<true, true, false><<<(int)((waves + 7) / 8), T, 0, stream>>>(
            agg1B, Wr1T, xmB, Wo1T, b_rel1, movie, nullptr, N_M, DD, HH);
    }
    // 6) conv2: userx = ReLU(agg2@Wrel2 + x_d@Wroot2 + b2)   [N_D x 256]
    {
        long waves = (long)(N_D / 16) * (HH / 64);
        k_gemm_wmma<true, true, false><<<(int)((waves + 7) / 8), T, 0, stream>>>(
            agg2B, Wr2T, xdB, Wo2T, b_rel2, userx, nullptr, N_D, DD, HH);
    }

    // 7) conv3 aggregation over movie features (256-d)
    k_scatter_add<false, HH><<<blk((long)NE * (HH / 4)), T, 0, stream>>>(movie, md_src, md_dst, nullptr, agg3);
    k_cvt_bf16<<<blk((long)N_D * HH / 4), T, 0, stream>>>(agg3,  agg3B,  N_D * HH / 4);
    k_cvt_bf16<<<blk((long)N_D * HH / 4), T, 0, stream>>>(userx, userxB, N_D * HH / 4);

    // 8) conv3: ux2 = ReLU(agg3@Wrel3 + userx@Wroot3 + b3)   [N_D x 256], bf16 out
    {
        long waves = (long)(N_D / 16) * (HH / 64);
        k_gemm_wmma<true, true, true><<<(int)((waves + 7) / 8), T, 0, stream>>>(
            agg3B, Wr3T, userxB, Wo3T, b_rel3, nullptr, ux2B, N_D, HH, HH);
    }
    // 9) final: out = ux2@Wlin + b_lin                        [N_D x 128], f32 out
    {
        long waves = (long)(N_D / 16) * (OO / 64);
        k_gemm_wmma<false, false, false><<<(int)((waves + 7) / 8), T, 0, stream>>>(
            ux2B, WlT, nullptr, nullptr, b_lin, (float*)d_out, nullptr, N_D, HH, OO);
    }
}
